// BiDAF_44014824849700
// MI455X (gfx1250) — matrix operand order repaired and checked by reading, not compiled, not run on hardware
//
#include <hip/hip_runtime.h>
#include <hip/hip_bf16.h>

// ---------------------------------------------------------------------------
// CDNA5 (gfx1250) BiDAF. bf16 WMMA (v_wmma_f32_16x16x32_bf16) everywhere,
// f32 accumulate. All B operands pre-swizzled to fragment-native layout
// [K/32][N/16][32 lanes][16 halves] so every fragment load is 2x b128.
// ---------------------------------------------------------------------------

typedef __bf16 bf16;
typedef __attribute__((ext_vector_type(16))) __bf16 v16bf;
typedef __attribute__((ext_vector_type(8)))  float  v8f;

static __device__ inline v8f wmma_bf16(v16bf a, v16bf b, v8f c) {
  return __builtin_amdgcn_wmma_f32_16x16x32_bf16(
      false, a, false, b, (short)0, c, false, false);
}

// A fragment 16x32 (MxK) bf16 from row-major (ld multiple of 8, 16B-aligned
// base). Lane L<16: M=L, K {0..15}; L>=16: M=L-16, K {16..31}.
// Each lane: one contiguous 16-half run -> two 128-bit loads.
static __device__ inline v16bf load_frag_a(const bf16* p, int ld) {
  int lane = threadIdx.x & 31;
  int m = lane & 15;
  int kb = (lane >> 4) << 4;
  const bf16* r = p + (size_t)m * ld + kb;
  union { uint4 u[2]; v16bf v; } x;
  x.u[0] = *(const uint4*)(r);
  x.u[1] = *(const uint4*)(r + 8);
  return x.v;
}

// B fragment 32x16 from pre-swizzled block: 32 contiguous bytes per lane.
static __device__ inline v16bf load_frag_b_sw(const bf16* blk) {
  int lane = threadIdx.x & 31;
  const uint4* q = (const uint4*)(blk + lane * 16);
  union { uint4 u[2]; v16bf v; } x;
  x.u[0] = q[0];
  x.u[1] = q[1];
  return x.v;
}

// C/D fragment 16x16 f32: VGPR r -> M = r + 8*(lane>=16), N = lane&15.
static __device__ inline v8f load_frag_c(const float* p, int ld) {
  int lane = threadIdx.x & 31;
  int n = lane & 15;
  int mo = (lane >> 4) << 3;
  v8f c;
#pragma unroll
  for (int r = 0; r < 8; ++r) c[r] = p[(size_t)(mo + r) * ld + n];
  return c;
}

static __device__ inline void store_frag_c(float* p, int ld, v8f c) {
  int lane = threadIdx.x & 31;
  int n = lane & 15;
  int mo = (lane >> 4) << 3;
#pragma unroll
  for (int r = 0; r < 8; ++r) p[(size_t)(mo + r) * ld + n] = c[r];
}

static __device__ inline float sigm(float x) { return 1.0f / (1.0f + __expf(-x)); }

// ---------------------------------------------------------------------------
// Weight prep: W [J, I] f32 (row-major)  ->  swizzled W^T blocks:
// out[((kc*(J/16) + nt)*32 + lane)*16 + i] = W[j, k],
//   k = kc*32 + ((lane>>4)<<4) + i, j = nt*16 + (lane&15); zero-pad k >= I.
// ---------------------------------------------------------------------------
__global__ void k_tcast_sw(const float* __restrict__ in, bf16* __restrict__ out,
                           int J, int I, int Kpad) {
  size_t idx = (size_t)blockIdx.x * blockDim.x + threadIdx.x;
  if (idx >= (size_t)Kpad * J) return;
  int i    = (int)(idx & 15);
  int lane = (int)((idx >> 4) & 31);
  size_t blk = idx >> 9;
  int NT = J >> 4;
  int nt = (int)(blk % NT);
  int kc = (int)(blk / NT);
  int k = kc * 32 + ((lane >> 4) << 4) + i;
  int j = nt * 16 + (lane & 15);
  out[idx] = (k < I) ? (bf16)in[(size_t)j * I + k] : (bf16)0.0f;
}

__global__ void k_vadd(const float* a, const float* b, float* o, int n) {
  int i = blockIdx.x * blockDim.x + threadIdx.x;
  if (i < n) o[i] = a[i] + b[i];
}

// ---------------------------------------------------------------------------
// Embedding gather -> time-major bf16 [T, N, Kpad], zero-padded. ids: [N, T]
// ---------------------------------------------------------------------------
__global__ void k_gather(const int* __restrict__ ids, const float* __restrict__ emb,
                         bf16* __restrict__ out, int T, int Nb, int E, int Kpad) {
  int row = blockIdx.x;               // t*Nb + n
  int t = row / Nb, n = row % Nb;
  int tok = ids[(size_t)n * T + t];
  const float* src = emb + (size_t)tok * E;
  bf16* dst = out + (size_t)row * Kpad;
  for (int k = threadIdx.x; k < Kpad; k += blockDim.x)
    dst[k] = (k < E) ? (bf16)src[k] : (bf16)0.0f;
}

// ---------------------------------------------------------------------------
// Batched WMMA GEMM: C[M,Nn] = A[M,K] * Bsw[K,Nn] (+bias[Nn]).
// A bf16 row-major (staged through LDS); Bsw pre-swizzled blocks (L2-hot).
// Block 256 thr (8 waves), 64x64 tile; grid (M/64, Nn/64, batch).
// ---------------------------------------------------------------------------
__global__ void k_gemm(const bf16* __restrict__ A, const bf16* __restrict__ Bsw,
                       const float* __restrict__ bias, float* __restrict__ C,
                       int M, int Nn, int K, size_t sA, size_t sB, size_t sC) {
  __shared__ __align__(16) bf16 sAm[64 * 32];
  const bf16* Ab = A + sA * blockIdx.z;
  const bf16* Bb = Bsw + sB * blockIdx.z;
  float* Cb = C + sC * blockIdx.z;
  int bm = blockIdx.x * 64, bn = blockIdx.y * 64;
  int tid = threadIdx.x, wave = tid >> 5;
  int t0 = wave * 2;
  int tm = (t0 >> 2) * 16, tn0 = (t0 & 3) * 16;
  int NT = Nn >> 4;
  int nt0 = (bn + tn0) >> 4;
  v8f acc0 = {}, acc1 = {};
  int ra = tid >> 2, ka = (tid & 3) * 8;     // A staging: 64 rows x 32 k
  for (int k0 = 0; k0 < K; k0 += 32) {
    *(uint4*)&sAm[ra * 32 + ka] = *(const uint4*)&Ab[(size_t)(bm + ra) * K + k0 + ka];
    if (k0 + 32 < K)
      __builtin_prefetch(&Ab[(size_t)(bm + ra) * K + k0 + 32 + ka], 0, 1);
    __syncthreads();
    v16bf a = load_frag_a(sAm + tm * 32, 32);
    const bf16* bp = Bb + ((size_t)(k0 >> 5) * NT + nt0) * 512;
    v16bf b0 = load_frag_b_sw(bp);
    v16bf b1 = load_frag_b_sw(bp + 512);
    acc0 = wmma_bf16(a, b0, acc0);
    acc1 = wmma_bf16(a, b1, acc1);
    __syncthreads();
  }
  int lane = tid & 31;
  int nn = lane & 15, mo = (lane >> 4) << 3;
  float bb0 = bias ? bias[bn + tn0 + nn] : 0.0f;
  float bb1 = bias ? bias[bn + tn0 + 16 + nn] : 0.0f;
#pragma unroll
  for (int r = 0; r < 8; ++r) {
    Cb[(size_t)(bm + tm + mo + r) * Nn + bn + tn0 + nn]      = acc0[r] + bb0;
    Cb[(size_t)(bm + tm + mo + r) * Nn + bn + tn0 + 16 + nn] = acc1[r] + bb1;
  }
}

// ---------------------------------------------------------------------------
// LSTM scan. gx = x@Wih^T + (bih+bhh), time-major [T, Nb, 512]. Per step:
// gates += h@Whh^T (WMMA, Whh^T swizzled, L2-hot), then cell update.
// Grid (4, 2): x = 16-row batch chunk, y = direction. Block 256 (8 waves).
// Hout f32 [Nb, T, 256] (fwd -> cols 0..127, bwd -> 128..255).
// ---------------------------------------------------------------------------
__global__ void k_lstm_scan(const float* __restrict__ gxF, const float* __restrict__ gxB,
                            const bf16* __restrict__ WTF, const bf16* __restrict__ WTB,
                            float* __restrict__ Hout, int T, int Nb) {
  extern __shared__ char smem[];
  float* sg  = (float*)smem;                 // 16*512 gates
  float* sc  = sg + 16 * 512;                // 16*128 cell
  bf16*  shb = (bf16*)(sc + 16 * 128);       // 16*128 h (bf16)
  int dir = blockIdx.y;
  int nbase = blockIdx.x * 16;
  const float* gx = dir ? gxB : gxF;
  const bf16*  WT = dir ? WTB : WTF;         // swizzled [4 kc][32 nt][512]
  int outOff = dir ? 128 : 0;
  int tid = threadIdx.x, wave = tid >> 5;

  for (int i = tid; i < 16 * 128; i += 256) { sc[i] = 0.0f; shb[i] = (bf16)0.0f; }
  __syncthreads();

  for (int s = 0; s < T; ++s) {
    int t = dir ? (T - 1 - s) : s;
    const float* g = gx + ((size_t)t * Nb + nbase) * 512;
    v16bf a0 = load_frag_a(shb + 0,  128);
    v16bf a1 = load_frag_a(shb + 32, 128);
    v16bf a2 = load_frag_a(shb + 64, 128);
    v16bf a3 = load_frag_a(shb + 96, 128);
#pragma unroll
    for (int j = 0; j < 4; ++j) {
      int nt = wave * 4 + j;
      int tn = nt * 16;
      v8f acc = load_frag_c(g + tn, 512);
      acc = wmma_bf16(a0, load_frag_b_sw(WT + (size_t)(0 * 32 + nt) * 512), acc);
      acc = wmma_bf16(a1, load_frag_b_sw(WT + (size_t)(1 * 32 + nt) * 512), acc);
      acc = wmma_bf16(a2, load_frag_b_sw(WT + (size_t)(2 * 32 + nt) * 512), acc);
      acc = wmma_bf16(a3, load_frag_b_sw(WT + (size_t)(3 * 32 + nt) * 512), acc);
      store_frag_c(sg + tn, 512, acc);
    }
    __syncthreads();
    for (int u = tid; u < 16 * 128; u += 256) {
      int n = u >> 7, h = u & 127;
      const float* gr = sg + n * 512;
      float iv = gr[h], fv = gr[128 + h], gv = gr[256 + h], ov = gr[384 + h];
      float cn = sigm(fv) * sc[u] + sigm(iv) * tanhf(gv);
      float hv = sigm(ov) * tanhf(cn);
      sc[u] = cn;
      shb[u] = (bf16)hv;
      Hout[((size_t)(nbase + n) * T + t) * 256 + outOff + h] = hv;
    }
    __syncthreads();
  }
}

// ---------------------------------------------------------------------------
// Attention prep (context): Hcw = Hc@wc, Acq = bf16(Hc * wcq). Wave per row.
// ---------------------------------------------------------------------------
__global__ void k_prep_c(const float* __restrict__ Hc, const float* __restrict__ wc,
                         const float* __restrict__ wcq, float* __restrict__ Hcw,
                         bf16* __restrict__ Acq) {
  int wave = threadIdx.x >> 5, lane = threadIdx.x & 31;
  size_t row = (size_t)blockIdx.x * 8 + wave;      // n*C + c
  const float* h = Hc + row * 256;
  float acc = 0.0f;
  for (int f = lane; f < 256; f += 32) {
    float v = h[f];
    acc += v * wc[f];
    Acq[row * 256 + f] = (bf16)(v * wcq[f]);
  }
#pragma unroll
  for (int m = 16; m; m >>= 1) acc += __shfl_xor(acc, m, 32);
  if (!lane) Hcw[row] = acc;
}

// Query prep: Hqw = Hq@wc, plus swizzled B forms:
//   HqS: scores B  [K=256(f), N=64(q)]  per batch (stride 256*64)
//   Hqc: c2q   B  [K=64(q),  N=256(f)] per batch (stride 64*256)
__global__ void k_prep_q(const float* __restrict__ Hq, const float* __restrict__ wc,
                         float* __restrict__ Hqw, bf16* __restrict__ HqS,
                         bf16* __restrict__ Hqc) {
  int wave = threadIdx.x >> 5, lane = threadIdx.x & 31;
  size_t row = (size_t)blockIdx.x * 8 + wave;      // n*64 + q
  int n = (int)(row >> 6), q = (int)(row & 63);
  const float* h = Hq + row * 256;
  bf16* hs = HqS + (size_t)n * 256 * 64;
  bf16* hc = Hqc + (size_t)n * 64 * 256;
  float acc = 0.0f;
  for (int f = lane; f < 256; f += 32) {
    float v = h[f];
    acc += v * wc[f];
    // scores: k=f, col=q
    size_t f1 = (((size_t)(f >> 5) * 4 + (q >> 4)) * 32 +
                 ((q & 15) + (((f >> 4) & 1) << 4))) * 16 + (f & 15);
    hs[f1] = (bf16)v;
    // c2q: k=q, col=f
    size_t f2 = (((size_t)(q >> 5) * 16 + (f >> 4)) * 32 +
                 ((f & 15) + (((q >> 4) & 1) << 4))) * 16 + (q & 15);
    hc[f2] = (bf16)v;
  }
#pragma unroll
  for (int m = 16; m; m >>= 1) acc += __shfl_xor(acc, m, 32);
  if (!lane) Hqw[row] = acc;
}

// ---------------------------------------------------------------------------
// S tile (64 c x 64 q) via WMMA; + column term Hqw; row softmax over Q.
// (Row term Hcw is softmax-invariant; folded into Smax only.)
// ---------------------------------------------------------------------------
__global__ void k_attn_scores(const bf16* __restrict__ Acq, const bf16* __restrict__ HqS,
                              const float* __restrict__ Hcw, const float* __restrict__ Hqw,
                              bf16* __restrict__ P, float* __restrict__ Smax, int C) {
  __shared__ float sS[64 * 66];
  int n = blockIdx.y, c0 = blockIdx.x * 64;
  const bf16* A = Acq + ((size_t)n * C + c0) * 256;    // [64,256] row-major
  const bf16* B = HqS + (size_t)n * 256 * 64;          // swizzled [8 kc][4 nt][512]
  int wave = threadIdx.x >> 5, lane = threadIdx.x & 31;
  int t0 = wave * 2;
  int tm = (t0 >> 2) * 16, tn0 = (t0 & 3) * 16;
  int nt0 = tn0 >> 4;
  v8f acc0 = {}, acc1 = {};
#pragma unroll
  for (int kc = 0; kc < 8; ++kc) {
    v16bf a = load_frag_a(A + (size_t)tm * 256 + kc * 32, 256);
    const bf16* bp = B + ((size_t)kc * 4 + nt0) * 512;
    acc0 = wmma_bf16(a, load_frag_b_sw(bp), acc0);
    acc1 = wmma_bf16(a, load_frag_b_sw(bp + 512), acc1);
  }
  store_frag_c(&sS[tm * 66 + tn0], 66, acc0);
  store_frag_c(&sS[tm * 66 + tn0 + 16], 66, acc1);
  __syncthreads();
  float hq0 = Hqw[(size_t)n * 64 + lane];
  float hq1 = Hqw[(size_t)n * 64 + lane + 32];
  for (int r = wave * 8; r < wave * 8 + 8; ++r) {
    float v0 = sS[r * 66 + lane] + hq0;
    float v1 = sS[r * 66 + lane + 32] + hq1;
    float mx = fmaxf(v0, v1);
#pragma unroll
    for (int m = 16; m; m >>= 1) mx = fmaxf(mx, __shfl_xor(mx, m, 32));
    float e0 = __expf(v0 - mx), e1 = __expf(v1 - mx);
    float sm = e0 + e1;
#pragma unroll
    for (int m = 16; m; m >>= 1) sm += __shfl_xor(sm, m, 32);
    float inv = 1.0f / sm;
    size_t base = ((size_t)n * C + c0 + r) * 64;
    P[base + lane]      = (bf16)(e0 * inv);
    P[base + lane + 32] = (bf16)(e1 * inv);
    if (!lane) Smax[(size_t)n * C + c0 + r] = mx + Hcw[(size_t)n * C + c0 + r];
  }
}

// q2c[n,f] = sum_c softmax_c(Smax[n])[c] * Hc[n,c,f].  One block per n.
__global__ void k_q2c(const float* __restrict__ Smax, const float* __restrict__ Hc,
                      float* __restrict__ q2c, int C) {
  __shared__ float sw[512];
  __shared__ float red[256];
  int n = blockIdx.x, tid = threadIdx.x;
  float a = Smax[(size_t)n * C + tid];
  float b = Smax[(size_t)n * C + 256 + tid];
  red[tid] = fmaxf(a, b);
  __syncthreads();
  for (int s = 128; s; s >>= 1) { if (tid < s) red[tid] = fmaxf(red[tid], red[tid + s]); __syncthreads(); }
  float mx = red[0];
  __syncthreads();
  float e0 = __expf(a - mx), e1 = __expf(b - mx);
  sw[tid] = e0; sw[256 + tid] = e1;
  red[tid] = e0 + e1;
  __syncthreads();
  for (int s = 128; s; s >>= 1) { if (tid < s) red[tid] += red[tid + s]; __syncthreads(); }
  float inv = 1.0f / red[0];
  float acc = 0.0f;
  for (int c = 0; c < C; ++c) acc += sw[c] * Hc[((size_t)n * C + c) * 256 + tid];
  q2c[(size_t)n * 256 + tid] = acc * inv;
}

// G = [Hc, c2q, Hc*c2q, Hc*q2c]: f32 [N*C,1024] + bf16 time-major [C,N,1024]
__global__ void k_build_G(const float* __restrict__ Hc, const float* __restrict__ c2q,
                          const float* __restrict__ q2c, float* __restrict__ G,
                          bf16* __restrict__ Gtm, int Nb, int C) {
  size_t idx = (size_t)blockIdx.x * blockDim.x + threadIdx.x;  // N*C*256
  int f = (int)(idx & 255);
  size_t row = idx >> 8;
  int n = (int)(row / C), c = (int)(row % C);
  float hc = Hc[idx];
  float cq = c2q[idx];
  float q2 = q2c[(size_t)n * 256 + f];
  size_t gb = row * 1024;
  size_t tb = ((size_t)c * Nb + n) * 1024;
  float p1 = hc * cq, p2 = hc * q2;
  G[gb + f]       = hc; Gtm[tb + f]       = (bf16)hc;
  G[gb + 256 + f] = cq; Gtm[tb + 256 + f] = (bf16)cq;
  G[gb + 512 + f] = p1; Gtm[tb + 512 + f] = (bf16)p1;
  G[gb + 768 + f] = p2; Gtm[tb + 768 + f] = (bf16)p2;
}

// f32 [Nb, T, 256] -> bf16 time-major [T, Nb, 256]
__global__ void k_cast_tm(const float* __restrict__ in, bf16* __restrict__ out,
                          int Nb, int T) {
  size_t idx = (size_t)blockIdx.x * blockDim.x + threadIdx.x;
  int f = (int)(idx & 255);
  size_t row = idx >> 8;
  int n = (int)(row / T), t = (int)(row % T);
  out[((size_t)t * Nb + n) * 256 + f] = (bf16)in[idx];
}

// start/end heads: out[row,0] = [G,M]@Ws + bs ; out[row,1] = [G,M2]@We + be
__global__ void k_heads(const float* __restrict__ G, const float* __restrict__ M,
                        const float* __restrict__ M2, const float* __restrict__ Ws,
                        const float* __restrict__ We, const float* __restrict__ bs,
                        const float* __restrict__ be, float* __restrict__ out) {
  int wave = threadIdx.x >> 5, lane = threadIdx.x & 31;
  size_t row = (size_t)blockIdx.x * 8 + wave;     // n*C + c
  const float* g  = G  + row * 1024;
  const float* m  = M  + row * 256;
  const float* m2 = M2 + row * 256;
  float s = 0.0f, e = 0.0f;
  for (int k = lane; k < 1024; k += 32) { float gv = g[k]; s += gv * Ws[k]; e += gv * We[k]; }
  for (int k = lane; k < 256;  k += 32) { s += m[k] * Ws[1024 + k]; e += m2[k] * We[1024 + k]; }
#pragma unroll
  for (int msk = 16; msk; msk >>= 1) { s += __shfl_xor(s, msk, 32); e += __shfl_xor(e, msk, 32); }
  if (!lane) { out[row * 2 + 0] = s + bs[0]; out[row * 2 + 1] = e + be[0]; }
}

// ---------------------------------------------------------------------------
// Host orchestration
// ---------------------------------------------------------------------------
extern "C" void kernel_launch(void* const* d_in, const int* in_sizes, int n_in,
                              void* d_out, int out_size, void* d_ws, size_t ws_size,
                              hipStream_t stream) {
  (void)in_sizes; (void)n_in; (void)out_size; (void)ws_size;
  const int N = 64, C = 512, Q = 64, E = 300;
  const int G4 = 512;          // 4*D gate width
  const int KE = 320;          // E padded to 32

  // jax tree: dict keys sorted; 'b' before 'f'; within lstm: Whh, Wih, bhh, bih
  const int*   ctx_ids = (const int*)d_in[0];
  const int*   qry_ids = (const int*)d_in[1];
  const float* emb     = (const float*)d_in[2];
  const float* We = (const float*)d_in[3];
  const float* Ws = (const float*)d_in[4];
  struct LW { const float *Whh, *Wih, *bhh, *bih; };
  auto getl = [&](int b0)->LW {
    return LW{ (const float*)d_in[b0], (const float*)d_in[b0+1],
               (const float*)d_in[b0+2], (const float*)d_in[b0+3] };
  };
  LW add_b  = getl(5),  add_f  = getl(9);
  LW ans0_b = getl(13), ans0_f = getl(17);
  LW ans1_b = getl(21), ans1_f = getl(25);
  const float* be = (const float*)d_in[29];
  const float* bs = (const float*)d_in[30];
  LW ctx_b = getl(31), ctx_f = getl(35);
  LW qry_b = getl(39), qry_f = getl(43);
  const float* wc  = (const float*)d_in[47];
  const float* wcq = (const float*)d_in[48];

  char* wsb = (char*)d_ws;
  size_t off = 0;
  auto alloc = [&](size_t bytes)->void* {
    void* p = wsb + off; off += (bytes + 255) & ~(size_t)255; return p;
  };
  struct Layer { LW f, b; int Kpad; bf16 *wTf, *wTb, *hTf, *hTb; float *bsf, *bsb; };
  Layer L[5] = {
    { ctx_f,  ctx_b,  KE,   0,0,0,0,0,0 },
    { qry_f,  qry_b,  KE,   0,0,0,0,0,0 },
    { ans0_f, ans0_b, 1024, 0,0,0,0,0,0 },
    { ans1_f, ans1_b, 256,  0,0,0,0,0,0 },
    { add_f,  add_b,  256,  0,0,0,0,0,0 },
  };
  for (int i = 0; i < 5; ++i) {
    L[i].wTf = (bf16*)alloc((size_t)L[i].Kpad * G4 * 2);
    L[i].wTb = (bf16*)alloc((size_t)L[i].Kpad * G4 * 2);
    L[i].hTf = (bf16*)alloc((size_t)128 * G4 * 2);
    L[i].hTb = (bf16*)alloc((size_t)128 * G4 * 2);
    L[i].bsf = (float*)alloc((size_t)G4 * 4);
    L[i].bsb = (float*)alloc((size_t)G4 * 4);
  }
  bf16*  ceb  = (bf16*)alloc((size_t)C * N * KE * 2);     // time-major ctx emb
  bf16*  qeb  = (bf16*)alloc((size_t)Q * N * KE * 2);
  float* gxF  = (float*)alloc((size_t)C * N * G4 * 4);    // reused per layer
  float* gxB  = (float*)alloc((size_t)C * N * G4 * 4);
  float* Hc   = (float*)alloc((size_t)N * C * 256 * 4);
  float* Hq   = (float*)alloc((size_t)N * Q * 256 * 4);
  bf16*  Acq  = (bf16*)alloc((size_t)N * C * 256 * 2);
  bf16*  HqS  = (bf16*)alloc((size_t)N * 256 * Q * 2);    // scores B, swizzled
  bf16*  Hqc  = (bf16*)alloc((size_t)N * Q * 256 * 2);    // c2q B, swizzled
  float* Hcw  = (float*)alloc((size_t)N * C * 4);
  float* Hqw  = (float*)alloc((size_t)N * Q * 4);
  bf16*  P    = (bf16*)alloc((size_t)N * C * Q * 2);
  float* Smax = (float*)alloc((size_t)N * C * 4);
  float* q2c  = (float*)alloc((size_t)N * 256 * 4);
  float* c2q  = (float*)alloc((size_t)N * C * 256 * 4);
  float* Gf   = (float*)alloc((size_t)N * C * 1024 * 4);
  bf16*  Gtm  = (bf16*)alloc((size_t)C * N * 1024 * 2);
  float* A0o  = (float*)alloc((size_t)N * C * 256 * 4);
  bf16*  A0tm = (bf16*)alloc((size_t)C * N * 256 * 2);
  float* Mo   = (float*)alloc((size_t)N * C * 256 * 4);
  bf16*  Mtm  = (bf16*)alloc((size_t)C * N * 256 * 2);
  float* M2o  = (float*)alloc((size_t)N * C * 256 * 4);

  // --- weight prep (swizzled W^T + fused bias) ---
  for (int i = 0; i < 5; ++i) {
    int I = (i < 2) ? E : ((i == 2) ? 1024 : 256);
    int Kpad = L[i].Kpad;
    int gT = (Kpad * G4 + 255) / 256;
    k_tcast_sw<<<gT, 256, 0, stream>>>(L[i].f.Wih, L[i].wTf, G4, I, Kpad);
    k_tcast_sw<<<gT, 256, 0, stream>>>(L[i].b.Wih, L[i].wTb, G4, I, Kpad);
    k_tcast_sw<<<(128 * G4 + 255) / 256, 256, 0, stream>>>(L[i].f.Whh, L[i].hTf, G4, 128, 128);
    k_tcast_sw<<<(128 * G4 + 255) / 256, 256, 0, stream>>>(L[i].b.Whh, L[i].hTb, G4, 128, 128);
    k_vadd<<<2, 256, 0, stream>>>(L[i].f.bih, L[i].f.bhh, L[i].bsf, G4);
    k_vadd<<<2, 256, 0, stream>>>(L[i].b.bih, L[i].b.bhh, L[i].bsb, G4);
  }

  // --- embedding gather ---
  k_gather<<<C * N, 128, 0, stream>>>(ctx_ids, emb, ceb, C, N, E, KE);
  k_gather<<<Q * N, 128, 0, stream>>>(qry_ids, emb, qeb, Q, N, E, KE);

  auto gemm = [&](const bf16* A, const bf16* B, const float* bias, float* Cm,
                  int M, int Nn, int K, size_t sA, size_t sB, size_t sC, int batch) {
    dim3 g(M / 64, Nn / 64, batch);
    k_gemm<<<g, 256, 0, stream>>>(A, B, bias, Cm, M, Nn, K, sA, sB, sC);
  };
  const size_t scanLds = (16 * 512 + 16 * 128) * 4 + 16 * 128 * 2;
  auto scan = [&](int li, float* Hout, int T) {
    k_lstm_scan<<<dim3(4, 2), 256, scanLds, stream>>>(gxF, gxB, L[li].hTf, L[li].hTb,
                                                      Hout, T, N);
  };

  // --- ctx BiLSTM ---
  gemm(ceb, L[0].wTf, L[0].bsf, gxF, C * N, G4, KE, 0, 0, 0, 1);
  gemm(ceb, L[0].wTb, L[0].bsb, gxB, C * N, G4, KE, 0, 0, 0, 1);
  scan(0, Hc, C);

  // --- qry BiLSTM ---
  gemm(qeb, L[1].wTf, L[1].bsf, gxF, Q * N, G4, KE, 0, 0, 0, 1);
  gemm(qeb, L[1].wTb, L[1].bsb, gxB, Q * N, G4, KE, 0, 0, 0, 1);
  scan(1, Hq, Q);

  // --- trilinear attention ---
  k_prep_c<<<N * C / 8, 256, 0, stream>>>(Hc, wc, wcq, Hcw, Acq);
  k_prep_q<<<N * Q / 8, 256, 0, stream>>>(Hq, wc, Hqw, HqS, Hqc);
  k_attn_scores<<<dim3(C / 64, N), 256, 0, stream>>>(Acq, HqS, Hcw, Hqw, P, Smax, C);
  k_q2c<<<N, 256, 0, stream>>>(Smax, Hc, q2c, C);
  gemm(P, Hqc, nullptr, c2q, C, 256, Q,
       (size_t)C * Q, (size_t)Q * 256, (size_t)C * 256, N);
  k_build_G<<<(N * C * 256) / 256, 256, 0, stream>>>(Hc, c2q, q2c, Gf, Gtm, N, C);

  // --- ans0 BiLSTM over G (K=1024) ---
  gemm(Gtm, L[2].wTf, L[2].bsf, gxF, C * N, G4, 1024, 0, 0, 0, 1);
  gemm(Gtm, L[2].wTb, L[2].bsb, gxB, C * N, G4, 1024, 0, 0, 0, 1);
  scan(2, A0o, C);
  k_cast_tm<<<(N * C * 256) / 256, 256, 0, stream>>>(A0o, A0tm, N, C);

  // --- ans1 BiLSTM (K=256) -> M ---
  gemm(A0tm, L[3].wTf, L[3].bsf, gxF, C * N, G4, 256, 0, 0, 0, 1);
  gemm(A0tm, L[3].wTb, L[3].bsb, gxB, C * N, G4, 256, 0, 0, 0, 1);
  scan(3, Mo, C);
  k_cast_tm<<<(N * C * 256) / 256, 256, 0, stream>>>(Mo, Mtm, N, C);

  // --- add BiLSTM (K=256) -> M2 ---
  gemm(Mtm, L[4].wTf, L[4].bsf, gxF, C * N, G4, 256, 0, 0, 0, 1);
  gemm(Mtm, L[4].wTb, L[4].bsb, gxB, C * N, G4, 256, 0, 0, 0, 1);
  scan(4, M2o, C);

  // --- output heads ---
  k_heads<<<N * C / 8, 256, 0, stream>>>(Gf, Mo, M2o, Ws, We, bs, be, (float*)d_out);
}
